// InvariantPointAttentionV41_11287174054524
// MI455X (gfx1250) — matrix operand order repaired and checked by reading, not compiled, not run on hardware
//
#include <hip/hip_runtime.h>

typedef __attribute__((ext_vector_type(16))) _Float16 v16h;
typedef __attribute__((ext_vector_type(8)))  float    v8f;
typedef __attribute__((ext_vector_type(4)))  float    v4f;

#define WMMA_F16(a, b, c) \
  __builtin_amdgcn_wmma_f32_16x16x32_f16(false, (a), false, (b), (short)0, (c), false, false)

// A fragment as loaded by this code (16x32, MxK): lane holds row M=lane&15;
// half hh -> K = (lane>>4)*16 + hh
// B fragment as loaded by this code (32x16, KxN): lane holds col N=lane&15;
// half hh -> K = (lane>>4)*16 + hh

// 8 consecutive-K floats; stride-1 path uses two aligned b128 loads.
// NT=true emits non-temporal (TH_NT) loads for read-once streams.
template <int SK, bool NT = false>
__device__ __forceinline__ void ld8(const float* __restrict__ p, float* d)
{
    if constexpr (SK == 1) {
        v4f u0, u1;
        if constexpr (NT) {
            u0 = __builtin_nontemporal_load(reinterpret_cast<const v4f*>(p));
            u1 = __builtin_nontemporal_load(reinterpret_cast<const v4f*>(p + 4));
        } else {
            u0 = *reinterpret_cast<const v4f*>(p);
            u1 = *reinterpret_cast<const v4f*>(p + 4);
        }
        d[0] = u0[0]; d[1] = u0[1]; d[2] = u0[2]; d[3] = u0[3];
        d[4] = u1[0]; d[5] = u1[1]; d[6] = u1[2]; d[7] = u1[3];
    } else {
#pragma unroll
        for (int t = 0; t < 8; ++t) {
            if constexpr (NT) d[t] = __builtin_nontemporal_load(p + (long long)t * SK);
            else              d[t] = p[(long long)t * SK];
        }
    }
}

// ---------------------------------------------------------------------------
// Generic WMMA GEMM, fully compile-time strides:
//   OUT[m*OSM + n*OSN] = alpha*sum_k A[m*ASM + k*ASK]*W[n*WSN + k*WSK]
//                        + bias[n] (+ OUT_old if ACC)
// One wave per 16x16 tile. M (gridDim.x*16) must be multiple of 16.
// N arbitrary (cols >= N computed on clamped row, never stored). K arbitrary.
// NTA: stream A with non-temporal loads (read-once data, bypass cache alloc).
// ---------------------------------------------------------------------------
template <int N, int K, int ASM, int ASK, int WSN, int WSK, int OSM, int OSN,
          bool BIAS, bool ACC, bool NTA = false>
__global__ __launch_bounds__(32) void gemm_wmma(
    const float* __restrict__ A, const float* __restrict__ W,
    const float* __restrict__ bias, float* __restrict__ OUT, float alpha)
{
    const int lane = threadIdx.x & 31;
    const int g    = lane >> 4;
    const int m0   = blockIdx.x << 4;
    const int n0   = blockIdx.y << 4;
    const int ml   = m0 + (lane & 15);
    const int nl   = n0 + (lane & 15);
    const int nw   = (nl < N) ? nl : (N - 1);      // clamp: cols >= N unstored
    const int kg8  = g << 3;
    const int kg16 = g << 4;

    const float* Ap = A + (long long)ml * ASM;
    const float* Wp = W + (long long)nw * WSN;

    v8f c = {0.f, 0.f, 0.f, 0.f, 0.f, 0.f, 0.f, 0.f};
    constexpr int KF = (K / 32) * 32;
    for (int k0 = 0; k0 < KF; k0 += 32) {
        float a0[8], a1[8], bb[16];
        ld8<ASK, NTA>(Ap + (long long)(k0 + kg16) * ASK, a0);
        ld8<ASK, NTA>(Ap + (long long)(k0 + kg16 + 8) * ASK, a1);
        ld8<WSK>(Wp + (long long)(k0 + kg16) * WSK, bb);
        ld8<WSK>(Wp + (long long)(k0 + kg16 + 8) * WSK, bb + 8);
        v16h af, bf;
#pragma unroll
        for (int t = 0; t < 8; ++t) { af[t] = (_Float16)a0[t]; af[t + 8] = (_Float16)a1[t]; }
#pragma unroll
        for (int t = 0; t < 16; ++t) bf[t] = (_Float16)bb[t];
        c = WMMA_F16(af, bf, c);
    }
    if constexpr (KF < K) {   // zero-padded tail, branch-free (clamp + select)
        v16h af, bf;
#pragma unroll
        for (int hh = 0; hh < 16; ++hh) {
            int ka  = KF + kg16 + hh;
            int kb  = KF + kg16 + hh;
            int kac = (ka < K) ? ka : (K - 1);
            int kbc = (kb < K) ? kb : (K - 1);
            float av = Ap[(long long)kac * ASK];
            float bv = Wp[(long long)kbc * WSK];
            af[hh] = (_Float16)((ka < K) ? av : 0.0f);
            bf[hh] = (_Float16)((kb < K) ? bv : 0.0f);
        }
        c = WMMA_F16(af, bf, c);
    }
    if (nl < N) {
        const float bval = BIAS ? bias[nl] : 0.0f;
#pragma unroll
        for (int r = 0; r < 8; ++r) {
            int m = m0 + r + kg8;
            long long idx = (long long)m * OSM + (long long)nl * OSN;
            float val = alpha * c[r] + bval;
            if constexpr (ACC) val += OUT[idx];
            OUT[idx] = val;
        }
    }
}

// ---------------------------------------------------------------------------
// Rigid transform: g = pts @ rot + trans   (768 residues x 48 points each)
// ---------------------------------------------------------------------------
__global__ void ipa_transform(const float* __restrict__ pts,
                              const float* __restrict__ rot,
                              const float* __restrict__ trans,
                              float* __restrict__ outg)
{
    int idx = blockIdx.x * blockDim.x + threadIdx.x;
    if (idx >= 768 * 48) return;
    int n = idx / 48, hp = idx % 48;
    const float* p = pts + n * 144 + hp * 3;
    const float* R = rot + n * 9;
    const float* t = trans + n * 3;
    float x = p[0], y = p[1], z = p[2];
    float* o = outg + n * 144 + hp * 3;
    o[0] = x * R[0] + y * R[3] + z * R[6] + t[0];
    o[1] = x * R[1] + y * R[4] + z * R[7] + t[1];
    o[2] = x * R[2] + y * R[5] + z * R[8] + t[2];
}

__global__ void ipa_sumsq(const float* __restrict__ gpts, float* __restrict__ s2)
{
    int idx = blockIdx.x * blockDim.x + threadIdx.x;
    if (idx >= 768 * 12) return;
    int n = idx / 12, h = idx % 12;
    const float* p = gpts + n * 144 + h * 12;
    float acc = 0.f;
#pragma unroll
    for (int i = 0; i < 12; ++i) acc += p[i] * p[i];
    s2[idx] = acc;
}

// ---------------------------------------------------------------------------
// Fused logits: logits[h,i,j] += scale * (q_i.k_j + qg_i.kg_j)
// (pair bias already in logits). Branch-free loads; two WMMAs per wave.
// ---------------------------------------------------------------------------
__global__ __launch_bounds__(32) void ipa_logits(
    const float* __restrict__ q, const float* __restrict__ k,
    const float* __restrict__ qg, const float* __restrict__ kg,
    float* __restrict__ logits, float scale)
{
    const int lane = threadIdx.x & 31;
    const int g    = lane >> 4;
    const int i0   = blockIdx.x << 4, j0 = blockIdx.y << 4, h = blockIdx.z;
    const int il   = i0 + (lane & 15), jl = j0 + (lane & 15);
    const int kg8  = g << 3, kg16 = g << 4;

    const float* qrow  = q  + (long long)il * 384 + h * 32;
    const float* krow  = k  + (long long)jl * 384 + h * 32;
    const float* qgrow = qg + (long long)il * 144 + h * 12;
    const float* kgrow = kg + (long long)jl * 144 + h * 12;

    v8f c = {0.f, 0.f, 0.f, 0.f, 0.f, 0.f, 0.f, 0.f};
    {   // scalar q.k, K=32 exactly: vectorized loads
        float a0[8], a1[8], bb[16];
        ld8<1>(qrow + kg16, a0);
        ld8<1>(qrow + kg16 + 8, a1);
        ld8<1>(krow + kg16, bb);
        ld8<1>(krow + kg16 + 8, bb + 8);
        v16h af, bf;
#pragma unroll
        for (int t = 0; t < 8; ++t) { af[t] = (_Float16)a0[t]; af[t + 8] = (_Float16)a1[t]; }
#pragma unroll
        for (int t = 0; t < 16; ++t) bf[t] = (_Float16)bb[t];
        c = WMMA_F16(af, bf, c);
    }
    {   // point qg.kg, K=12 zero-padded (clamp + select, no branches)
        v16h ag, bg;
#pragma unroll
        for (int hh = 0; hh < 16; ++hh) {
            int ka  = kg16 + hh;
            int kb  = kg16 + hh;
            float av = qgrow[(ka < 12) ? ka : 11];
            float bv = kgrow[(kb < 12) ? kb : 11];
            ag[hh] = (_Float16)((ka < 12) ? av : 0.0f);
            bg[hh] = (_Float16)((kb < 12) ? bv : 0.0f);
        }
        c = WMMA_F16(ag, bg, c);
    }
#pragma unroll
    for (int r = 0; r < 8; ++r) {
        int i = i0 + r + kg8;
        long long idx = (long long)h * 589824LL + (long long)i * 768 + jl;
        logits[idx] += scale * c[r];
    }
}

// ---------------------------------------------------------------------------
// Softmax over j; folds the j-dependent -0.5*scale*k2[j,h] term.  The
// row-constant q2[i,h] term is softmax-invariant and dropped (exact).
// ---------------------------------------------------------------------------
__global__ __launch_bounds__(256) void ipa_softmax(
    float* __restrict__ logits, const float* __restrict__ k2, float scale)
{
    __shared__ float red[256];
    const int i = blockIdx.x, h = blockIdx.y, tid = threadIdx.x;
    float* row = logits + (long long)h * 589824LL + (long long)i * 768;
    float vals[3];
    float mx = -3.402823466e38f;
#pragma unroll
    for (int c = 0; c < 3; ++c) {
        int j = tid + c * 256;
        float v = row[j] - 0.5f * scale * k2[j * 12 + h];
        vals[c] = v;
        mx = fmaxf(mx, v);
    }
    red[tid] = mx; __syncthreads();
    for (int st = 128; st > 0; st >>= 1) { if (tid < st) red[tid] = fmaxf(red[tid], red[tid + st]); __syncthreads(); }
    mx = red[0]; __syncthreads();
    float s = 0.f;
#pragma unroll
    for (int c = 0; c < 3; ++c) { vals[c] = __expf(vals[c] - mx); s += vals[c]; }
    red[tid] = s; __syncthreads();
    for (int st = 128; st > 0; st >>= 1) { if (tid < st) red[tid] += red[tid + st]; __syncthreads(); }
    float inv = 1.0f / red[0];
#pragma unroll
    for (int c = 0; c < 3; ++c) row[tid + c * 256] = vals[c] * inv;
}

// ---------------------------------------------------------------------------
// Fused attn apply: one A-fragment per K-step feeds 3 WMMAs:
//   weighted[i, h*32+0..31] = attn[h] @ v[:, h*32+0..31]   (two N-tiles)
//   pw[i, h*12+0..11]       = attn[h] @ vg[:, h*12+0..11]
// ---------------------------------------------------------------------------
__global__ __launch_bounds__(32) void ipa_attn_apply(
    const float* __restrict__ attn, const float* __restrict__ v,
    const float* __restrict__ vg, float* __restrict__ weighted,
    float* __restrict__ pw)
{
    const int lane = threadIdx.x & 31;
    const int g    = lane >> 4;
    const int i0   = blockIdx.x << 4;
    const int h    = blockIdx.y;
    const int il   = i0 + (lane & 15), nl = lane & 15;
    const int kg8  = g << 3, kg16 = g << 4;
    const int nvg  = (nl < 12) ? nl : 11;          // clamp: cols 12..15 unstored
    const float* arow = attn + (long long)h * 589824LL + (long long)il * 768;

    v8f c0 = {0.f,0.f,0.f,0.f,0.f,0.f,0.f,0.f};
    v8f c1 = {0.f,0.f,0.f,0.f,0.f,0.f,0.f,0.f};
    v8f c2 = {0.f,0.f,0.f,0.f,0.f,0.f,0.f,0.f};
    for (int k0 = 0; k0 < 768; k0 += 32) {
        float a0[8], a1[8];
        ld8<1>(arow + k0 + kg16, a0);
        ld8<1>(arow + k0 + kg16 + 8, a1);
        v16h af, b0, b1, b2;
#pragma unroll
        for (int t = 0; t < 8; ++t) { af[t] = (_Float16)a0[t]; af[t + 8] = (_Float16)a1[t]; }
        const float* vbase  = v  + (long long)(k0 + kg16) * 384 + h * 32 + nl;
        const float* vgbase = vg + (long long)(k0 + kg16) * 144 + h * 12 + nvg;
#pragma unroll
        for (int t = 0; t < 16; ++t) {
            b0[t] = (_Float16)vbase[(long long)t * 384];
            b1[t] = (_Float16)vbase[(long long)t * 384 + 16];
            b2[t] = (_Float16)vgbase[(long long)t * 144];
        }
        c0 = WMMA_F16(af, b0, c0);
        c1 = WMMA_F16(af, b1, c1);
        c2 = WMMA_F16(af, b2, c2);
    }
#pragma unroll
    for (int r = 0; r < 8; ++r) {
        int i = i0 + r + kg8;
        weighted[(long long)i * 384 + h * 32 + nl]      = c0[r];
        weighted[(long long)i * 384 + h * 32 + 16 + nl] = c1[r];
        if (nl < 12) pw[(long long)i * 144 + h * 12 + nl] = c2[r];
    }
}

// ---------------------------------------------------------------------------
// Residual + LayerNorm over C=384. 128 threads, 3 elements each.
// ---------------------------------------------------------------------------
__global__ __launch_bounds__(128) void ipa_layernorm(
    const float* __restrict__ single, const float* __restrict__ proj,
    const float* __restrict__ gam, const float* __restrict__ bet,
    float* __restrict__ out)
{
    __shared__ float red[128];
    const int n = blockIdx.x, tid = threadIdx.x;
    float x[3]; float s = 0.f;
#pragma unroll
    for (int i = 0; i < 3; ++i) {
        int c = tid + i * 128;
        x[i] = single[n * 384 + c] + proj[n * 384 + c];
        s += x[i];
    }
    red[tid] = s; __syncthreads();
    for (int st = 64; st > 0; st >>= 1) { if (tid < st) red[tid] += red[tid + st]; __syncthreads(); }
    float mu = red[0] * (1.0f / 384.0f); __syncthreads();
    float s2 = 0.f;
#pragma unroll
    for (int i = 0; i < 3; ++i) { float d = x[i] - mu; s2 += d * d; }
    red[tid] = s2; __syncthreads();
    for (int st = 64; st > 0; st >>= 1) { if (tid < st) red[tid] += red[tid + st]; __syncthreads(); }
    float inv = rsqrtf(red[0] * (1.0f / 384.0f) + 1e-5f);
#pragma unroll
    for (int i = 0; i < 3; ++i) {
        int c = tid + i * 128;
        out[n * 384 + c] = (x[i] - mu) * inv * gam[c] + bet[c];
    }
}

// ===========================================================================
extern "C" void kernel_launch(void* const* d_in, const int* in_sizes, int n_in,
                              void* d_out, int out_size, void* d_ws, size_t ws_size,
                              hipStream_t stream)
{
    const float* single = (const float*)d_in[0];
    const float* pair   = (const float*)d_in[1];
    const float* rot    = (const float*)d_in[2];
    const float* trans  = (const float*)d_in[3];
    const float* q_w  = (const float*)d_in[4];  const float* q_b  = (const float*)d_in[5];
    const float* k_w  = (const float*)d_in[6];  const float* k_b  = (const float*)d_in[7];
    const float* v_w  = (const float*)d_in[8];  const float* v_b  = (const float*)d_in[9];
    const float* pb_w = (const float*)d_in[10]; const float* pb_b = (const float*)d_in[11];
    const float* qp_w = (const float*)d_in[12]; const float* qp_b = (const float*)d_in[13];
    const float* kp_w = (const float*)d_in[14]; const float* kp_b = (const float*)d_in[15];
    const float* vp_w = (const float*)d_in[16]; const float* vp_b = (const float*)d_in[17];
    const float* pt_w = (const float*)d_in[18]; const float* pt_b = (const float*)d_in[19];
    const float* out_w = (const float*)d_in[20]; const float* out_b = (const float*)d_in[21];
    const float* ln_g = (const float*)d_in[22]; const float* ln_b = (const float*)d_in[23];

    const float scale = 0.17677669529663687f;   // 32^-0.5

    float* ws = (float*)d_ws;
    float* q        = ws;              ws += 768 * 384;
    float* kbuf     = ws;              ws += 768 * 384;
    float* vbuf     = ws;              ws += 768 * 384;
    float* qp       = ws;              ws += 768 * 144;
    float* kp       = ws;              ws += 768 * 144;
    float* vp       = ws;              ws += 768 * 144;
    float* qg       = ws;              ws += 768 * 144;
    float* kg       = ws;              ws += 768 * 144;
    float* vg       = ws;              ws += 768 * 144;
    float* k2       = ws;              ws += 768 * 12;
    float* logits   = ws;              ws += 12 * 768 * 768;
    float* weighted = ws;              ws += 768 * 384;
    float* pw       = ws;              ws += 768 * 144;
    float* proj     = ws;              ws += 768 * 384;

    // ---- scalar q/k/v projections: [768,384] = single @ W^T + b ----
    gemm_wmma<384, 384, 384, 1, 384, 1, 384, 1, true, false>
        <<<dim3(48, 24), 32, 0, stream>>>(single, q_w, q_b, q, 1.0f);
    gemm_wmma<384, 384, 384, 1, 384, 1, 384, 1, true, false>
        <<<dim3(48, 24), 32, 0, stream>>>(single, k_w, k_b, kbuf, 1.0f);
    gemm_wmma<384, 384, 384, 1, 384, 1, 384, 1, true, false>
        <<<dim3(48, 24), 32, 0, stream>>>(single, v_w, v_b, vbuf, 1.0f);

    // ---- point projections: [768,144] ----
    gemm_wmma<144, 384, 384, 1, 384, 1, 144, 1, true, false>
        <<<dim3(48, 9), 32, 0, stream>>>(single, qp_w, qp_b, qp, 1.0f);
    gemm_wmma<144, 384, 384, 1, 384, 1, 144, 1, true, false>
        <<<dim3(48, 9), 32, 0, stream>>>(single, kp_w, kp_b, kp, 1.0f);
    gemm_wmma<144, 384, 384, 1, 384, 1, 144, 1, true, false>
        <<<dim3(48, 9), 32, 0, stream>>>(single, vp_w, vp_b, vp, 1.0f);

    // ---- rigid transforms + squared norms ----
    ipa_transform<<<144, 256, 0, stream>>>(qp, rot, trans, qg);
    ipa_transform<<<144, 256, 0, stream>>>(kp, rot, trans, kg);
    ipa_transform<<<144, 256, 0, stream>>>(vp, rot, trans, vg);
    ipa_sumsq<<<36, 256, 0, stream>>>(kg, k2);

    // ---- pair bias -> logits[h,i,j] ----
    // pair is 302 MB read-once (> 192 MB L2): stream it non-temporally so the
    // 28 MB logits buffer stays L2-resident for the logits/softmax/attn chain.
    gemm_wmma<12, 128, 128, 1, 128, 1, 1, 589824, true, false, true /*NTA*/>
        <<<dim3(36864, 1), 32, 0, stream>>>(pair, pb_w, pb_b, logits, 1.0f);

    // ---- scalar + point logits fused, accumulate onto pair bias ----
    ipa_logits<<<dim3(48, 48, 12), 32, 0, stream>>>(q, kbuf, qg, kg, logits, scale);

    // ---- softmax ----
    ipa_softmax<<<dim3(768, 12), 256, 0, stream>>>(logits, k2, scale);

    // ---- attn @ v  and  attn @ vg, fused (attn read once) ----
    ipa_attn_apply<<<dim3(48, 12), 32, 0, stream>>>(logits, vbuf, vg, weighted, pw);

    // ---- point projection, accumulated into weighted (K=144 padded tail) ----
    gemm_wmma<384, 144, 144, 1, 144, 1, 384, 1, true, true>
        <<<dim3(48, 24), 32, 0, stream>>>(pw, pt_w, pt_b, weighted, 1.0f);

    // ---- output projection ----
    gemm_wmma<384, 384, 384, 1, 384, 1, 384, 1, true, false>
        <<<dim3(48, 24), 32, 0, stream>>>(weighted, out_w, out_b, proj, 1.0f);

    // ---- residual + layernorm -> d_out ----
    ipa_layernorm<<<768, 128, 0, stream>>>(single, proj, ln_g, ln_b, (float*)d_out);
}
